// Trans_Global_SSG_33655363732081
// MI455X (gfx1250) — compile-verified, matrix operand order checked by reading, not run on hardware
//
#include <hip/hip_runtime.h>
#include <hip/hip_bf16.h>
#include <math.h>

// ---------------------------------------------------------------------------
// Sizes from the reference
// ---------------------------------------------------------------------------
enum {
  Bb = 8, Nn = 4096, Ss = 1024, Kk = 32, ICH = 64,
  C1 = 128, C2 = 128, C3 = 256, Cc = 256, Hh = 8,
  BS = Bb * Ss,              // 8192 tokens
  Mg = Bb * Ss * Kk,         // 262144 grouped points
  K0P = 96                   // 67 input channels padded to multiple of 32
};
#define RADIUS2 (0.2f * 0.2f)
#define EPSV 1e-5f

typedef __attribute__((ext_vector_type(16))) _Float16 v16h;
typedef __attribute__((ext_vector_type(8)))  _Float16 v8h;
typedef __attribute__((ext_vector_type(8)))  float    v8f;

// Fragment halves are two contiguous 8xf16 runs; concatenate them.
__device__ __forceinline__ v16h cat8(v8h lo, v8h hi) {
  return __builtin_shufflevector(lo, hi, 0, 1, 2, 3, 4, 5, 6, 7,
                                 8, 9, 10, 11, 12, 13, 14, 15);
}

// ---------------------------------------------------------------------------
// 1) Farthest point sampling: one block per batch (sequential argmax chain)
// ---------------------------------------------------------------------------
__global__ __launch_bounds__(256)
void fps_kernel(const float* __restrict__ xyz, int* __restrict__ fidx) {
  __shared__ float sdist[Nn];
  __shared__ float sval[256];
  __shared__ int   sidx[256];
  __shared__ int   sfar;
  const int b = blockIdx.x, t = threadIdx.x;
  const float* px = xyz + (size_t)b * 3 * Nn;
  const float* py = px + Nn;
  const float* pz = py + Nn;
  for (int n = t; n < Nn; n += 256) sdist[n] = 1e10f;
  if (t == 0) sfar = 0;
  __syncthreads();
  for (int it = 0; it < Ss; ++it) {
    const int far = sfar;
    if (t == 0) fidx[b * Ss + it] = far;
    const float cx = px[far], cy = py[far], cz = pz[far];
    float bestv = -1.f; int besti = 0;
    for (int n = t; n < Nn; n += 256) {
      float dx = px[n] - cx, dy = py[n] - cy, dz = pz[n] - cz;
      float nd = fminf(sdist[n], dx * dx + dy * dy + dz * dz);
      sdist[n] = nd;
      if (nd > bestv) { bestv = nd; besti = n; }
    }
    sval[t] = bestv; sidx[t] = besti;
    __syncthreads();
    for (int st = 128; st > 0; st >>= 1) {
      if (t < st) {
        float ov = sval[t + st]; int oi = sidx[t + st];
        if (ov > sval[t] || (ov == sval[t] && oi < sidx[t])) { sval[t] = ov; sidx[t] = oi; }
      }
      __syncthreads();
    }
    if (t == 0) sfar = sidx[0];
    __syncthreads();
  }
}

// ---------------------------------------------------------------------------
// 2) Gather sampled centers; also emit first output (B,3,S)
// ---------------------------------------------------------------------------
__global__ void newxyz_kernel(const float* __restrict__ xyz, const int* __restrict__ fidx,
                              float* __restrict__ nxyz, float* __restrict__ out_xyz) {
  int idx = blockIdx.x * blockDim.x + threadIdx.x;
  if (idx >= BS) return;
  int b = idx / Ss, s = idx % Ss;
  int n = fidx[idx];
  for (int d = 0; d < 3; ++d) {
    float v = xyz[(size_t)b * 3 * Nn + (size_t)d * Nn + n];
    nxyz[(size_t)idx * 3 + d] = v;
    out_xyz[(size_t)b * 3 * Ss + (size_t)d * Ss + s] = v;
  }
}

// ---------------------------------------------------------------------------
// 3) Ball query: first K in-radius indices in ascending order, pad with first
// ---------------------------------------------------------------------------
__global__ void ballquery_kernel(const float* __restrict__ xyz, const float* __restrict__ nxyz,
                                 int* __restrict__ gidx) {
  int idx = blockIdx.x * blockDim.x + threadIdx.x;
  if (idx >= BS) return;
  int b = idx / Ss;
  const float* px = xyz + (size_t)b * 3 * Nn;
  const float* py = px + Nn;
  const float* pz = py + Nn;
  float cx = nxyz[idx * 3], cy = nxyz[idx * 3 + 1], cz = nxyz[idx * 3 + 2];
  int* gr = gidx + (size_t)idx * Kk;
  int cnt = 0;
  for (int n = 0; n < Nn && cnt < Kk; ++n) {
    float dx = px[n] - cx, dy = py[n] - cy, dz = pz[n] - cz;
    if (dx * dx + dy * dy + dz * dz <= RADIUS2) gr[cnt++] = n;
  }
  int f = (cnt > 0) ? gr[0] : 0;
  for (; cnt < Kk; ++cnt) gr[cnt] = f;
}

// ---------------------------------------------------------------------------
// 4) Build grouped features directly in f16, K padded to 96 with zeros
// ---------------------------------------------------------------------------
__global__ void group_kernel(const float* __restrict__ xyz, const float* __restrict__ points,
                             const float* __restrict__ nxyz, const int* __restrict__ gidx,
                             _Float16* __restrict__ X0) {
  int m = blockIdx.x * blockDim.x + threadIdx.x;
  if (m >= Mg) return;
  int bs = m / Kk;
  int b = bs / Ss;
  int gi = gidx[m];
  _Float16* row = X0 + (size_t)m * K0P;
  const float* px = xyz + (size_t)b * 3 * Nn;
  row[0] = (_Float16)(px[gi]          - nxyz[bs * 3 + 0]);
  row[1] = (_Float16)(px[Nn + gi]     - nxyz[bs * 3 + 1]);
  row[2] = (_Float16)(px[2 * Nn + gi] - nxyz[bs * 3 + 2]);
  const float* pp = points + (size_t)b * ICH * Nn;
  for (int c = 0; c < ICH; ++c) row[3 + c] = (_Float16)pp[(size_t)c * Nn + gi];
  for (int c = 3 + ICH; c < K0P; ++c) row[c] = (_Float16)0.f;
}

// ---------------------------------------------------------------------------
// 5) Weight convert: W[N,Kd] f32 -> W16[N,Kpad] f16, zero tail
// ---------------------------------------------------------------------------
__global__ void wcvt_kernel(const float* __restrict__ W, _Float16* __restrict__ W16,
                            int N, int Kd, int Kpad) {
  int idx = blockIdx.x * blockDim.x + threadIdx.x;
  if (idx >= N * Kpad) return;
  int n = idx / Kpad, k = idx % Kpad;
  W16[idx] = (k < Kd) ? (_Float16)W[(size_t)n * Kd + k] : (_Float16)0.f;
}

// ---------------------------------------------------------------------------
// 6) f16 WMMA GEMM: one wave computes a 16x64 tile (4 WMMAs / K-step).
//    All operands pre-converted f16, K multiple of 32 -> pure b128 loads.
//    Output: fp32 (optional +res) or f16; optional bias/relu.
// ---------------------------------------------------------------------------
__global__ __launch_bounds__(256)
void gemm_wmma16(const _Float16* __restrict__ A, int lda,
                 const _Float16* __restrict__ W, int ldw,
                 const float* __restrict__ bias,
                 const float* __restrict__ res, int ldr,
                 int out_relu,
                 float* __restrict__ Y32, _Float16* __restrict__ Y16, int ldy,
                 int M, int N, int Kd) {
  int wave = (blockIdx.x * blockDim.x + threadIdx.x) >> 5;
  int lane = threadIdx.x & 31;
  int nTn = N >> 6;                       // 64-wide N tiles
  int total = (M >> 4) * nTn;
  if (wave >= total) return;              // wave-uniform, EXEC stays all-ones
  int tm = wave / nTn, tn = wave % nTn;
  int hv = lane >> 4, r = lane & 15;

  const v8h* Ar  = (const v8h*)(A + (size_t)(tm * 16 + r) * lda);
  const v8h* Wr0 = (const v8h*)(W + (size_t)(tn * 64 +  0 + r) * ldw);
  const v8h* Wr1 = (const v8h*)(W + (size_t)(tn * 64 + 16 + r) * ldw);
  const v8h* Wr2 = (const v8h*)(W + (size_t)(tn * 64 + 32 + r) * ldw);
  const v8h* Wr3 = (const v8h*)(W + (size_t)(tn * 64 + 48 + r) * ldw);

  v8f acc[4] = {{}, {}, {}, {}};
  int kv = hv;                            // v8h index of this lane's low run
  for (int k0 = 0; k0 < Kd; k0 += 32, kv += 4) {
    v16h af = cat8(Ar[kv],  Ar[kv + 2]);
    v16h b0 = cat8(Wr0[kv], Wr0[kv + 2]);
    v16h b1 = cat8(Wr1[kv], Wr1[kv + 2]);
    v16h b2 = cat8(Wr2[kv], Wr2[kv + 2]);
    v16h b3 = cat8(Wr3[kv], Wr3[kv + 2]);
    acc[0] = __builtin_amdgcn_wmma_f32_16x16x32_f16(false, af, false, b0, (short)0, acc[0], false, false);
    acc[1] = __builtin_amdgcn_wmma_f32_16x16x32_f16(false, af, false, b1, (short)0, acc[1], false, false);
    acc[2] = __builtin_amdgcn_wmma_f32_16x16x32_f16(false, af, false, b2, (short)0, acc[2], false, false);
    acc[3] = __builtin_amdgcn_wmma_f32_16x16x32_f16(false, af, false, b3, (short)0, acc[3], false, false);
  }
#pragma unroll
  for (int nt = 0; nt < 4; ++nt) {
#pragma unroll
    for (int i = 0; i < 8; ++i) {
      int orow = tm * 16 + hv * 8 + i;
      int ocol = tn * 64 + nt * 16 + r;
      float v = acc[nt][i];
      if (bias) v += bias[ocol];
      if (res)  v += res[(size_t)orow * ldr + ocol];
      if (out_relu) v = fmaxf(v, 0.f);
      if (Y32) Y32[(size_t)orow * ldy + ocol] = v;
      if (Y16) Y16[(size_t)orow * ldy + ocol] = (_Float16)v;
    }
  }
}

// ---------------------------------------------------------------------------
// 7) BatchNorm stats: deterministic two-stage per-channel reduction
// ---------------------------------------------------------------------------
__global__ __launch_bounds__(256)
void bn_stats_kernel(const float* __restrict__ Y, int M, int C,
                     float* __restrict__ psum, float* __restrict__ psumsq) {
  int c = blockIdx.x;
  __shared__ float ss[256], ss2[256];
  float s = 0.f, s2 = 0.f;
  for (int m = blockIdx.y * blockDim.x + threadIdx.x; m < M; m += gridDim.y * blockDim.x) {
    float v = Y[(size_t)m * C + c];
    s += v; s2 += v * v;
  }
  ss[threadIdx.x] = s; ss2[threadIdx.x] = s2;
  __syncthreads();
  for (int st = 128; st > 0; st >>= 1) {
    if (threadIdx.x < st) { ss[threadIdx.x] += ss[threadIdx.x + st]; ss2[threadIdx.x] += ss2[threadIdx.x + st]; }
    __syncthreads();
  }
  if (threadIdx.x == 0) {
    psum[(size_t)c * gridDim.y + blockIdx.y]   = ss[0];
    psumsq[(size_t)c * gridDim.y + blockIdx.y] = ss2[0];
  }
}

__global__ void bn_finalize_kernel(const float* __restrict__ psum, const float* __restrict__ psumsq,
                                   int nParts, const float* __restrict__ g, const float* __restrict__ bb,
                                   float* __restrict__ scale, float* __restrict__ shift,
                                   int C, float invM) {
  int c = blockIdx.x * blockDim.x + threadIdx.x;
  if (c >= C) return;
  float s = 0.f, s2 = 0.f;
  for (int p = 0; p < nParts; ++p) { s += psum[(size_t)c * nParts + p]; s2 += psumsq[(size_t)c * nParts + p]; }
  float mean = s * invM;
  float var  = s2 * invM - mean * mean;
  float sc   = g[c] * rsqrtf(var + EPSV);
  scale[c] = sc;
  shift[c] = bb[c] - mean * sc;
}

// ---------------------------------------------------------------------------
// 8) BN+ReLU apply + f16 convert (feeds next GEMM)
// ---------------------------------------------------------------------------
__global__ void actcvt_kernel(const float* __restrict__ Y, const float* __restrict__ sc,
                              const float* __restrict__ sh, _Float16* __restrict__ X,
                              int total, int C) {
  int idx = blockIdx.x * blockDim.x + threadIdx.x;
  if (idx >= total) return;
  int k = idx % C;
  X[idx] = (_Float16)fmaxf(Y[idx] * sc[k] + sh[k], 0.f);
}

// ---------------------------------------------------------------------------
// 9) Apply final BN+ReLU and max-pool over K neighbors -> tok[BS, C3]
// ---------------------------------------------------------------------------
__global__ void maxpool_kernel(const float* __restrict__ Y3, const float* __restrict__ scale,
                               const float* __restrict__ shift, float* __restrict__ tok) {
  int idx = blockIdx.x * blockDim.x + threadIdx.x;
  if (idx >= BS * C3) return;
  int c = idx % C3, bs = idx / C3;
  float sc = scale[c], sh = shift[c];
  const float* base = Y3 + (size_t)bs * Kk * C3 + c;
  float mx = -1e30f;
  for (int k = 0; k < Kk; ++k) mx = fmaxf(mx, fmaxf(sc * base[(size_t)k * C3] + sh, 0.f));
  tok[idx] = mx;
}

// ---------------------------------------------------------------------------
// 10) Positional encoding: h = tok + nxyz @ pos_w^T + pos_b   (K=3: VALU)
// ---------------------------------------------------------------------------
__global__ void posadd_kernel(const float* __restrict__ tok, const float* __restrict__ nxyz,
                              const float* __restrict__ pw, const float* __restrict__ pb,
                              float* __restrict__ h) {
  int idx = blockIdx.x * blockDim.x + threadIdx.x;
  if (idx >= BS * Cc) return;
  int c = idx % Cc, bs = idx / Cc;
  h[idx] = tok[idx] + pw[c * 3] * nxyz[bs * 3] + pw[c * 3 + 1] * nxyz[bs * 3 + 1]
                    + pw[c * 3 + 2] * nxyz[bs * 3 + 2] + pb[c];
}

// ---------------------------------------------------------------------------
// 11) LayerNorm (fp32 in, f16 out): one wave per row
// ---------------------------------------------------------------------------
__global__ __launch_bounds__(256)
void layernorm_kernel(const float* __restrict__ x, const float* __restrict__ g,
                      const float* __restrict__ b, _Float16* __restrict__ y, int rows) {
  int wave = (blockIdx.x * blockDim.x + threadIdx.x) >> 5;
  int lane = threadIdx.x & 31;
  if (wave >= rows) return;
  const float* xr = x + (size_t)wave * Cc;
  float s = 0.f, s2 = 0.f;
  for (int c = lane; c < Cc; c += 32) { float v = xr[c]; s += v; s2 += v * v; }
#pragma unroll
  for (int off = 1; off < 32; off <<= 1) { s += __shfl_xor(s, off, 32); s2 += __shfl_xor(s2, off, 32); }
  float mean = s * (1.f / Cc);
  float inv  = rsqrtf(s2 * (1.f / Cc) - mean * mean + EPSV);
  for (int c = lane; c < Cc; c += 32)
    y[(size_t)wave * Cc + c] = (_Float16)((xr[c] - mean) * inv * g[c] + b[c]);
}

// ---------------------------------------------------------------------------
// 12) V transpose for attention: v16[bs,c] -> Vt[(b,h,n), s]  (f16)
// ---------------------------------------------------------------------------
__global__ void vtrans_kernel(const _Float16* __restrict__ v16, _Float16* __restrict__ Vt) {
  int idx = blockIdx.x * blockDim.x + threadIdx.x;
  if (idx >= BS * Cc) return;
  int c = idx % Cc, bs = idx / Cc;
  int b = bs / Ss, s = bs % Ss;
  Vt[((size_t)(b * Hh + (c >> 5)) * 32 + (c & 31)) * Ss + s] = v16[idx];
}

// ---------------------------------------------------------------------------
// 13) Flash attention: one wave per (b, head, 16-query tile). All f16 inputs,
//     contiguous b128 fragment loads, LDS bounce for P re-layout.
// ---------------------------------------------------------------------------
__global__ __launch_bounds__(32)
void attention_kernel(const _Float16* __restrict__ q, const _Float16* __restrict__ k,
                      const _Float16* __restrict__ Vt, _Float16* __restrict__ o) {
  const int b = blockIdx.z, h = blockIdx.y, qt = blockIdx.x;
  const int lane = threadIdx.x & 31, hv = lane >> 4, r = lane & 15;
  const float sc = rsqrtf(32.f);          // 1/sqrt(hd)
  __shared__ __align__(16) _Float16 pbuf[16][32];

  const int qrow = qt * 16 + r;
  const v8h* qr = (const v8h*)(q + ((size_t)(b * Ss + qrow)) * Cc + h * 32);
  v16h aq = cat8(qr[hv], qr[2 + hv]);

  const _Float16* vt0 = Vt + ((size_t)(b * Hh + h) * 32 + r) * Ss;       // n = r
  const _Float16* vt1 = vt0 + (size_t)16 * Ss;                           // n = 16+r

  float mrow[8], lrow[8];
  v8f o0 = {}, o1 = {};
#pragma unroll
  for (int i = 0; i < 8; ++i) { mrow[i] = -1e30f; lrow[i] = 0.f; }

  for (int kc = 0; kc < Ss; kc += 32) {
    const v8h* kr0 = (const v8h*)(k + ((size_t)(b * Ss + kc + r)) * Cc + h * 32);
    const v8h* kr1 = (const v8h*)(k + ((size_t)(b * Ss + kc + 16 + r)) * Cc + h * 32);
    v16h bk0 = cat8(kr0[hv], kr0[2 + hv]);
    v16h bk1 = cat8(kr1[hv], kr1[2 + hv]);
    v8f z = {};
    v8f s0 = __builtin_amdgcn_wmma_f32_16x16x32_f16(false, aq, false, bk0, (short)0, z, false, false);
    v8f s1 = __builtin_amdgcn_wmma_f32_16x16x32_f16(false, aq, false, bk1, (short)0, z, false, false);

#pragma unroll
    for (int i = 0; i < 8; ++i) {
      float a0 = s0[i] * sc, a1 = s1[i] * sc;
      float tmax = fmaxf(a0, a1);
#pragma unroll
      for (int off = 1; off < 16; off <<= 1) tmax = fmaxf(tmax, __shfl_xor(tmax, off, 16));
      float mn = fmaxf(mrow[i], tmax);
      float corr = __expf(mrow[i] - mn);
      float p0 = __expf(a0 - mn), p1 = __expf(a1 - mn);
      float rs = p0 + p1;
#pragma unroll
      for (int off = 1; off < 16; off <<= 1) rs += __shfl_xor(rs, off, 16);
      lrow[i] = lrow[i] * corr + rs;
      mrow[i] = mn;
      o0[i] *= corr; o1[i] *= corr;
      pbuf[hv * 8 + i][r]      = (_Float16)p0;
      pbuf[hv * 8 + i][16 + r] = (_Float16)p1;
    }
    __syncthreads();
    const v8h* pr  = (const v8h*)&pbuf[r][0];
    const v8h* v0p = (const v8h*)(vt0 + kc);
    const v8h* v1p = (const v8h*)(vt1 + kc);
    v16h pa  = cat8(pr[hv],  pr[2 + hv]);
    v16h bv0 = cat8(v0p[hv], v0p[2 + hv]);
    v16h bv1 = cat8(v1p[hv], v1p[2 + hv]);
    o0 = __builtin_amdgcn_wmma_f32_16x16x32_f16(false, pa, false, bv0, (short)0, o0, false, false);
    o1 = __builtin_amdgcn_wmma_f32_16x16x32_f16(false, pa, false, bv1, (short)0, o1, false, false);
    __syncthreads();
  }
#pragma unroll
  for (int i = 0; i < 8; ++i) {
    int orow = qt * 16 + hv * 8 + i;
    float inv = 1.f / lrow[i];
    o[((size_t)(b * Ss + orow)) * Cc + h * 32 + r]      = (_Float16)(o0[i] * inv);
    o[((size_t)(b * Ss + orow)) * Cc + h * 32 + 16 + r] = (_Float16)(o1[i] * inv);
  }
}

// ---------------------------------------------------------------------------
// 14) Pack final h into output (B, C, S)
// ---------------------------------------------------------------------------
__global__ void packh_kernel(const float* __restrict__ hf, float* __restrict__ out) {
  int idx = blockIdx.x * blockDim.x + threadIdx.x;
  if (idx >= BS * Cc) return;
  int c = idx % Cc, bs = idx / Cc;
  int b = bs / Ss, s = bs % Ss;
  out[(size_t)b * Cc * Ss + (size_t)c * Ss + s] = hf[idx];
}

// ---------------------------------------------------------------------------
// Host side
// ---------------------------------------------------------------------------
static inline void launch_gemm(const _Float16* A, int lda, const _Float16* W, int ldw,
                               const float* bias, const float* res, int ldr, int relu,
                               float* Y32, _Float16* Y16, int ldy,
                               int M, int N, int Kd, hipStream_t s) {
  int waves = (M / 16) * (N / 64);
  int blocks = (waves * 32 + 255) / 256;
  gemm_wmma16<<<blocks, 256, 0, s>>>(A, lda, W, ldw, bias, res, ldr, relu, Y32, Y16, ldy, M, N, Kd);
}

extern "C" void kernel_launch(void* const* d_in, const int* in_sizes, int n_in,
                              void* d_out, int out_size, void* d_ws, size_t ws_size,
                              hipStream_t stream) {
  (void)in_sizes; (void)n_in; (void)out_size; (void)ws_size;
  const float* xyz     = (const float*)d_in[0];
  const float* points  = (const float*)d_in[1];
  const float* conv_w[3] = { (const float*)d_in[2], (const float*)d_in[3], (const float*)d_in[4] };
  const float* conv_b[3] = { (const float*)d_in[5], (const float*)d_in[6], (const float*)d_in[7] };
  const float* bn_g[3]   = { (const float*)d_in[8], (const float*)d_in[9], (const float*)d_in[10] };
  const float* bn_b[3]   = { (const float*)d_in[11], (const float*)d_in[12], (const float*)d_in[13] };
  const float* pos_w = (const float*)d_in[14];
  const float* pos_b = (const float*)d_in[15];
  const float* Wq = (const float*)d_in[16];
  const float* Wkp = (const float*)d_in[17];
  const float* Wv = (const float*)d_in[18];
  const float* Wo = (const float*)d_in[19];
  const float* bq = (const float*)d_in[20];
  const float* bk = (const float*)d_in[21];
  const float* bv = (const float*)d_in[22];
  const float* bo = (const float*)d_in[23];
  const float* ln_g[2] = { (const float*)d_in[24], (const float*)d_in[25] };
  const float* ln_b[2] = { (const float*)d_in[26], (const float*)d_in[27] };
  const float* W1 = (const float*)d_in[28];
  const float* W2 = (const float*)d_in[29];
  const float* b1 = (const float*)d_in[30];
  const float* b2 = (const float*)d_in[31];

  float* out_xyz = (float*)d_out;
  float* out_h   = (float*)d_out + (size_t)Bb * 3 * Ss;

  char* ws = (char*)d_ws;
  size_t off = 0;
  auto alloc = [&](size_t bytes) -> void* {
    void* p = ws + off;
    off += (bytes + 255) & ~(size_t)255;
    return p;
  };
  int*       fidx = (int*)      alloc((size_t)BS * 4);
  int*       gidx = (int*)      alloc((size_t)BS * Kk * 4);
  float*     nxyz = (float*)    alloc((size_t)BS * 3 * 4);
  _Float16*  X0   = (_Float16*) alloc((size_t)Mg * K0P * 2);
  float*     Y1   = (float*)    alloc((size_t)Mg * C1 * 4);
  _Float16*  X1   = (_Float16*) alloc((size_t)Mg * C1 * 2);
  float*     Y2   = (float*)    alloc((size_t)Mg * C2 * 4);
  _Float16*  X2   = (_Float16*) alloc((size_t)Mg * C2 * 2);
  float*     Y3   = (float*)    alloc((size_t)Mg * C3 * 4);
  const int NP = 64;
  float* psum = (float*)alloc((size_t)C3 * NP * 4);
  float* psq  = (float*)alloc((size_t)C3 * NP * 4);
  float* bsc  = (float*)alloc((size_t)C3 * 4);
  float* bsh  = (float*)alloc((size_t)C3 * 4);
  // f16 weights (padded K)
  _Float16* w0_16 = (_Float16*)alloc((size_t)C1 * K0P * 2);
  _Float16* w1_16 = (_Float16*)alloc((size_t)C2 * C1 * 2);
  _Float16* w2_16 = (_Float16*)alloc((size_t)C3 * C2 * 2);
  _Float16* wq16  = (_Float16*)alloc((size_t)Cc * Cc * 2);
  _Float16* wk16  = (_Float16*)alloc((size_t)Cc * Cc * 2);
  _Float16* wv16  = (_Float16*)alloc((size_t)Cc * Cc * 2);
  _Float16* wo16  = (_Float16*)alloc((size_t)Cc * Cc * 2);
  _Float16* wf1   = (_Float16*)alloc((size_t)4 * Cc * Cc * 2);
  _Float16* wf2   = (_Float16*)alloc((size_t)Cc * 4 * Cc * 2);
  // transformer activations
  float*    tok  = (float*)   alloc((size_t)BS * Cc * 4);
  float*    hbuf = (float*)   alloc((size_t)BS * Cc * 4);
  _Float16* hn   = (_Float16*)alloc((size_t)BS * Cc * 2);
  _Float16* q16  = (_Float16*)alloc((size_t)BS * Cc * 2);
  _Float16* k16  = (_Float16*)alloc((size_t)BS * Cc * 2);
  _Float16* v16b = (_Float16*)alloc((size_t)BS * Cc * 2);
  _Float16* Vt   = (_Float16*)alloc((size_t)BS * Cc * 2);
  _Float16* ob16 = (_Float16*)alloc((size_t)BS * Cc * 2);
  float*    h2   = (float*)   alloc((size_t)BS * Cc * 4);
  _Float16* hn2  = (_Float16*)alloc((size_t)BS * Cc * 2);
  _Float16* F16  = (_Float16*)alloc((size_t)BS * 4 * Cc * 2);
  float*    hfin = (float*)   alloc((size_t)BS * Cc * 4);

  // ---- weight conversions ----
  auto wcvt = [&](const float* W, _Float16* W16, int N, int Kd, int Kpad) {
    wcvt_kernel<<<(N * Kpad + 255) / 256, 256, 0, stream>>>(W, W16, N, Kd, Kpad);
  };
  wcvt(conv_w[0], w0_16, C1, ICH + 3, K0P);
  wcvt(conv_w[1], w1_16, C2, C1, C1);
  wcvt(conv_w[2], w2_16, C3, C2, C2);
  wcvt(Wq, wq16, Cc, Cc, Cc);
  wcvt(Wkp, wk16, Cc, Cc, Cc);
  wcvt(Wv, wv16, Cc, Cc, Cc);
  wcvt(Wo, wo16, Cc, Cc, Cc);
  wcvt(W1, wf1, 4 * Cc, Cc, Cc);
  wcvt(W2, wf2, Cc, 4 * Cc, 4 * Cc);

  // ---- geometry ----
  fps_kernel<<<Bb, 256, 0, stream>>>(xyz, fidx);
  newxyz_kernel<<<(BS + 255) / 256, 256, 0, stream>>>(xyz, fidx, nxyz, out_xyz);
  ballquery_kernel<<<(BS + 255) / 256, 256, 0, stream>>>(xyz, nxyz, gidx);
  group_kernel<<<(Mg + 255) / 256, 256, 0, stream>>>(xyz, points, nxyz, gidx, X0);

  const float invMg = 1.f / (float)Mg;
  // ---- conv MLP ----
  launch_gemm(X0, K0P, w0_16, K0P, conv_b[0], nullptr, 0, 0, Y1, nullptr, C1, Mg, C1, K0P, stream);
  bn_stats_kernel<<<dim3(C1, NP), 256, 0, stream>>>(Y1, Mg, C1, psum, psq);
  bn_finalize_kernel<<<1, C1, 0, stream>>>(psum, psq, NP, bn_g[0], bn_b[0], bsc, bsh, C1, invMg);
  actcvt_kernel<<<(Mg * C1 + 255) / 256, 256, 0, stream>>>(Y1, bsc, bsh, X1, Mg * C1, C1);

  launch_gemm(X1, C1, w1_16, C1, conv_b[1], nullptr, 0, 0, Y2, nullptr, C2, Mg, C2, C1, stream);
  bn_stats_kernel<<<dim3(C2, NP), 256, 0, stream>>>(Y2, Mg, C2, psum, psq);
  bn_finalize_kernel<<<1, C2, 0, stream>>>(psum, psq, NP, bn_g[1], bn_b[1], bsc, bsh, C2, invMg);
  actcvt_kernel<<<(Mg * C2 + 255) / 256, 256, 0, stream>>>(Y2, bsc, bsh, X2, Mg * C2, C2);

  launch_gemm(X2, C2, w2_16, C2, conv_b[2], nullptr, 0, 0, Y3, nullptr, C3, Mg, C3, C2, stream);
  bn_stats_kernel<<<dim3(C3, NP), 256, 0, stream>>>(Y3, Mg, C3, psum, psq);
  bn_finalize_kernel<<<1, C3, 0, stream>>>(psum, psq, NP, bn_g[2], bn_b[2], bsc, bsh, C3, invMg);

  maxpool_kernel<<<(BS * C3 + 255) / 256, 256, 0, stream>>>(Y3, bsc, bsh, tok);
  posadd_kernel<<<(BS * Cc + 255) / 256, 256, 0, stream>>>(tok, nxyz, pos_w, pos_b, hbuf);

  // ---- transformer ----
  layernorm_kernel<<<(BS * 32 + 255) / 256, 256, 0, stream>>>(hbuf, ln_g[0], ln_b[0], hn, BS);

  launch_gemm(hn, Cc, wq16, Cc, bq, nullptr, 0, 0, nullptr, q16, Cc, BS, Cc, Cc, stream);
  launch_gemm(hn, Cc, wk16, Cc, bk, nullptr, 0, 0, nullptr, k16, Cc, BS, Cc, Cc, stream);
  launch_gemm(hn, Cc, wv16, Cc, bv, nullptr, 0, 0, nullptr, v16b, Cc, BS, Cc, Cc, stream);
  vtrans_kernel<<<(BS * Cc + 255) / 256, 256, 0, stream>>>(v16b, Vt);

  attention_kernel<<<dim3(Ss / 16, Hh, Bb), 32, 0, stream>>>(q16, k16, Vt, ob16);

  launch_gemm(ob16, Cc, wo16, Cc, bo, hbuf, Cc, 0, h2, nullptr, Cc, BS, Cc, Cc, stream);
  layernorm_kernel<<<(BS * 32 + 255) / 256, 256, 0, stream>>>(h2, ln_g[1], ln_b[1], hn2, BS);

  launch_gemm(hn2, Cc, wf1, Cc, b1, nullptr, 0, 1, nullptr, F16, 4 * Cc, BS, 4 * Cc, Cc, stream);
  launch_gemm(F16, 4 * Cc, wf2, 4 * Cc, b2, h2, Cc, 0, hfin, nullptr, Cc, BS, Cc, 4 * Cc, stream);

  packh_kernel<<<(BS * Cc + 255) / 256, 256, 0, stream>>>(hfin, out_h);
}
